// DepST_RNN_56160992362627
// MI455X (gfx1250) — matrix operand order repaired and checked by reading, not compiled, not run on hardware
//
#include <hip/hip_runtime.h>

// Problem constants (from reference)
#define B_ 8
#define L_ 8
#define E_ 128
#define N_ 1024
#define NODE_ 256
#define DEP_ 128
#define R_ 40
#define F_ (NODE_ + DEP_)   // 384

typedef __attribute__((ext_vector_type(16))) _Float16 v16h;
typedef __attribute__((ext_vector_type(8)))  float    v8f;

union HV { v16h h; uint4 u[2]; };

// ---------------------------------------------------------------------------
// Setup kernels (run once per launch)
// ---------------------------------------------------------------------------

// out[b,n,0:256] = context (static), out[b,n,256:384] = child (init 0)
__global__ __launch_bounds__(256) void k_init_out(const float* __restrict__ ctx,
                                                  float* __restrict__ out) {
    int idx = blockIdx.x * 256 + threadIdx.x;          // < B*N*F
    int bn = idx / F_, f = idx - bn * F_;
    out[idx] = (f < NODE_) ? ctx[bn * NODE_ + f] : 0.0f;
}

__global__ __launch_bounds__(256) void k_init_ws(float* __restrict__ sums,
                                                 float* __restrict__ counts) {
    int idx = blockIdx.x * 256 + threadIdx.x;          // < B*N*DEP + B*N
    if (idx < B_ * N_ * DEP_) sums[idx] = 0.0f;
    else                      counts[idx - B_ * N_ * DEP_] = 0.0f;
}

__global__ __launch_bounds__(256) void k_convw(const float* __restrict__ w,
                                               _Float16* __restrict__ wh) {
    int idx = blockIdx.x * 256 + threadIdx.x;          // < R*DEP*F
    wh[idx] = (_Float16)w[idx];
}

// ---------------------------------------------------------------------------
// Per-layer kernel 1: counting-sort the 1024 edges of this layer by relation,
// padding each relation group to a multiple of 16 (dummy slots: mask=0).
// Also accumulates the scatter-mean denominator (sum of masks per head).
// ---------------------------------------------------------------------------
__global__ __launch_bounds__(1024) void k_bucket(const int* __restrict__ heads,
                                                 const int* __restrict__ tails,
                                                 const int* __restrict__ rels,
                                                 const float* __restrict__ mask,
                                                 int layer,
                                                 float* __restrict__ counts,
                                                 int* __restrict__ slot_tail,
                                                 int* __restrict__ slot_head,
                                                 float* __restrict__ slot_mask,
                                                 int* __restrict__ tile_rel,
                                                 int* __restrict__ tile_base,
                                                 int* __restrict__ ntiles) {
    __shared__ int cnt[R_];
    __shared__ int base[R_];
    int tid = threadIdx.x;
    if (tid < R_) cnt[tid] = 0;
    // clear padded slot arrays (2048 slots; dummies keep mask=0, node 0)
    for (int s = tid; s < 2048; s += 1024) {
        slot_mask[s] = 0.0f; slot_head[s] = 0; slot_tail[s] = 0;
    }
    __syncthreads();

    int b = tid >> 7, e = tid & (E_ - 1);
    int idx = (b * L_ + layer) * E_ + e;
    int r = rels[idx], h = heads[idx], t = tails[idx];
    float m = mask[idx];
    int rank = atomicAdd(&cnt[r], 1);
    __syncthreads();

    if (tid == 0) {
        int run = 0, nt = 0;
        for (int rr = 0; rr < R_; ++rr) {
            base[rr] = run;
            int pc = (cnt[rr] + 15) & ~15;
            for (int j = 0; j < pc; j += 16) {
                tile_rel[nt] = rr; tile_base[nt] = run + j; ++nt;
            }
            run += pc;
        }
        ntiles[0] = nt;                                 // worst case 101 <= 128
    }
    __syncthreads();

    int s = base[r] + rank;
    slot_tail[s] = b * N_ + t;
    slot_head[s] = b * N_ + h;
    slot_mask[s] = m;
    atomicAdd(&counts[b * N_ + h], m);
}

// ---------------------------------------------------------------------------
// Per-layer kernel 2: one workgroup per (relation, 16-edge tile).
// 8 waves; wave w computes dep rows [16w,16w+16) of W[rel] @ FG_tile via
// v_wmma_f32_16x16x32_f16, K = 384 in 12 steps. Scatter-add via f32 atomics.
// ---------------------------------------------------------------------------
__global__ __launch_bounds__(256) void k_gather(const float* __restrict__ ctx,
                                                const float* __restrict__ outc,
                                                const _Float16* __restrict__ wh,
                                                const int* __restrict__ slot_tail,
                                                const int* __restrict__ slot_head,
                                                const float* __restrict__ slot_mask,
                                                const int* __restrict__ tile_rel,
                                                const int* __restrict__ tile_base,
                                                const int* __restrict__ ntiles,
                                                float* __restrict__ sums) {
    __shared__ _Float16 fg[16 * F_];     // 16 edges x 384 halves = 12.3 KB
    __shared__ float smask[16];
    __shared__ int   shead[16];

    int t = blockIdx.x;
    if (t >= ntiles[0]) return;          // uniform early-exit
    int rel = tile_rel[t];
    int tb  = tile_base[t];
    int tid = threadIdx.x;

    if (tid < 16) { smask[tid] = slot_mask[tb + tid]; shead[tid] = slot_head[tb + tid]; }
    // stage gathered features (concat(context, child)[tail]) as f16 in LDS
    for (int i = tid; i < 16 * F_; i += 256) {
        int edge = i / F_, k = i - edge * F_;
        int g = slot_tail[tb + edge];
        // child lives in the output buffer at [g*F_ + 256 + (k-256)] == [g*F_ + k]
        float v = (k < NODE_) ? ctx[g * NODE_ + k] : outc[(size_t)g * F_ + k];
        fg[i] = (_Float16)v;
    }
    __syncthreads();

    int lane = tid & 31;
    int wave = tid >> 5;                 // m-tile index 0..7
    int lo = lane & 15, hi = lane >> 4;

    // A operand: W rows. lane -> (M = lo, K-half = hi). 16x32 f16 per step.
    const _Float16* wrow = wh + ((size_t)rel * DEP_ + wave * 16 + lo) * F_;
    // B operand: fg columns. lane -> (N = lo edge, K = 8*hi..+7 and +16).
    const _Float16* frow = &fg[lo * F_];

    v8f acc = {};
#pragma unroll
    for (int ks = 0; ks < F_ / 32; ++ks) {
        HV a, bm;
        int ka = ks * 32 + hi * 8;                 // A: K = ka..ka+7 and +16
        a.u[0]  = *(const uint4*)(wrow + ka);
        a.u[1]  = *(const uint4*)(wrow + ka + 16);
        int kb = ks * 32 + hi * 8;                 // B: K = kb..kb+7 and +16
        bm.u[0] = *(const uint4*)(frow + kb);
        bm.u[1] = *(const uint4*)(frow + kb + 16);
        acc = __builtin_amdgcn_wmma_f32_16x16x32_f16(
                  false, a.h, false, bm.h, (short)0, acc, false, false);
    }

    // D layout: lane holds (N = lo, M = i + 8*hi). Mask then scatter-add.
    float m = smask[lo];
    int g = shead[lo];
    float* sp = sums + (size_t)g * DEP_ + wave * 16 + hi * 8;
#pragma unroll
    for (int i = 0; i < 8; ++i) atomicAdd(sp + i, acc[i] * m);
}

// ---------------------------------------------------------------------------
// Per-layer kernel 3: child = (count>0) ? sums / max(count,1) : old child.
// Also re-zeroes sums/counts for the next layer.
// ---------------------------------------------------------------------------
__global__ __launch_bounds__(256) void k_finalize(float* __restrict__ counts,
                                                  float* __restrict__ sums,
                                                  float* __restrict__ out) {
    int idx = blockIdx.x * 256 + threadIdx.x;      // < B*N*DEP
    int bn = idx >> 7, d = idx & (DEP_ - 1);
    float c = counts[bn];
    float s = sums[idx];
    __syncthreads();                               // all readers of counts[bn] are in-block
    if (c > 0.0f) out[(size_t)bn * F_ + NODE_ + d] = s / fmaxf(c, 1.0f);
    sums[idx] = 0.0f;
    if (d == 0) counts[bn] = 0.0f;
}

// ---------------------------------------------------------------------------
extern "C" void kernel_launch(void* const* d_in, const int* in_sizes, int n_in,
                              void* d_out, int out_size, void* d_ws, size_t ws_size,
                              hipStream_t stream) {
    (void)in_sizes; (void)n_in; (void)out_size; (void)ws_size;
    const float* ctx   = (const float*)d_in[0];
    const float* depW  = (const float*)d_in[1];
    const int*   heads = (const int*)d_in[2];
    const int*   tails = (const int*)d_in[3];
    const int*   rels  = (const int*)d_in[4];
    const float* mask  = (const float*)d_in[5];
    float* out = (float*)d_out;

    // Workspace layout (~8.2 MB)
    char* ws = (char*)d_ws;
    float*    sums      = (float*)(ws);                 // 4,194,304 B
    float*    counts    = (float*)(ws + 4194304);       //    32,768 B
    _Float16* wh        = (_Float16*)(ws + 4227072);    // 3,932,160 B
    int*      slot_tail = (int*)(ws + 8159232);         //     8,192 B
    int*      slot_head = (int*)(ws + 8167424);         //     8,192 B
    float*    slot_mask = (float*)(ws + 8175616);       //     8,192 B
    int*      tile_rel  = (int*)(ws + 8183808);         //       512 B
    int*      tile_base = (int*)(ws + 8184320);         //       512 B
    int*      ntiles    = (int*)(ws + 8184832);         //        16 B

    k_init_out<<<(B_ * N_ * F_) / 256, 256, 0, stream>>>(ctx, out);
    k_init_ws<<<(B_ * N_ * DEP_ + B_ * N_) / 256, 256, 0, stream>>>(sums, counts);
    k_convw<<<(R_ * DEP_ * F_) / 256, 256, 0, stream>>>(depW, wh);

    for (int l = 0; l < L_; ++l) {
        k_bucket<<<1, 1024, 0, stream>>>(heads, tails, rels, mask, l, counts,
                                         slot_tail, slot_head, slot_mask,
                                         tile_rel, tile_base, ntiles);
        k_gather<<<128, 256, 0, stream>>>(ctx, out, wh, slot_tail, slot_head,
                                          slot_mask, tile_rel, tile_base,
                                          ntiles, sums);
        k_finalize<<<(B_ * N_ * DEP_) / 256, 256, 0, stream>>>(counts, sums, out);
    }
}
